// DosNet_54563264528558
// MI455X (gfx1250) — matrix operand order repaired and checked by reading, not compiled, not run on hardware
//
#include <hip/hip_runtime.h>
#include <hip/hip_bf16.h>

// ---- CDNA5 WMMA types ----
typedef __attribute__((ext_vector_type(16))) __bf16 v16bf;
typedef __attribute__((ext_vector_type(8)))  __bf16 v8bf;
typedef __attribute__((ext_vector_type(4)))  __bf16 v4bf;
typedef __attribute__((ext_vector_type(4)))  float  v4f;
typedef __attribute__((ext_vector_type(8)))  float  v8f;
typedef __attribute__((ext_vector_type(4)))  int    v4i;

#define N_NODES 50000
#define N_EDGES 1000000
#define NSC     64
#define NVC     32
#define INV_DEG 0.22360679774997896f   /* 1/sqrt(E/N)=1/sqrt(20) */
#define EPSR    1e-9f

__device__ __forceinline__ float silu(float x){ return x / (1.0f + __expf(-x)); }
__device__ __forceinline__ __bf16 bfc(float x){ return (__bf16)x; }
__device__ __forceinline__ v8f zero8(){ v8f z = {0.f,0.f,0.f,0.f,0.f,0.f,0.f,0.f}; return z; }
__device__ __forceinline__ v16bf concat8(v8bf lo, v8bf hi){
  return __builtin_shufflevector(lo, hi, 0,1,2,3,4,5,6,7,8,9,10,11,12,13,14,15);
}

__device__ __forceinline__ v8f wmma_bf16(v16bf a, v16bf b, v8f c){
  // D = A(16x32 bf16) * B(32x16 bf16) + C(16x16 f32)
  return __builtin_amdgcn_wmma_f32_16x16x32_bf16(false, a, false, b, (short)0, c, false, false);
}

// A-fragment (16x32 bf16) from a row-major tile: two contiguous 16B loads per lane.
// lanes 0-15: row=lane, K={kb..kb+15}, kb=0; lanes 16-31: kb=16.
__device__ __forceinline__ v16bf load_afrag(const __bf16* base, int ld, int kbase){
  const int lane = threadIdx.x & 31;
  const __bf16* p = base + (lane & 15) * ld + kbase + ((lane < 16) ? 0 : 16);
  v8bf lo = *(const v8bf*)p;
  v8bf hi = *(const v8bf*)(p + 8);
  return concat8(lo, hi);
}

// B-fragment from pre-swizzled storage: 512-element block, 16 contiguous bf16 per lane.
__device__ __forceinline__ v16bf load_bfrag_sw(const __bf16* blk){
  const int lane = threadIdx.x & 31;
  const __bf16* p = blk + lane * 16;
  v8bf lo = *(const v8bf*)p;
  v8bf hi = *(const v8bf*)(p + 8);
  return concat8(lo, hi);
}

// ---------------- kernel 0: weights -> bf16, padded, fragment-swizzled ----------------
// swizzle: block (kc,nt) of 512; within block: lane*16+i holds
//   W[kc*32 + ((lane<16)?0:16) + i][nt*16 + (lane&15)]
__global__ void prep_weights(const float* We1, const float* We2, const float* Wr1,
                             const float* Wr2, const float* Ws, const float* Wv,
                             const float* Wos, const float* Wov,
                             __bf16* we1b, __bf16* we2b, __bf16* wosb, __bf16* wovb,
                             __bf16* wr1b, __bf16* wr2b, __bf16* wsvb){
  const int tid = threadIdx.x + blockIdx.x * blockDim.x;
  const int nth = blockDim.x * gridDim.x;
  for (int f = tid; f < 9216; f += nth){          // We1/We2/Wsv: 3x6 blocks (96x96 padded)
    int blk = f >> 9, w = f & 511, lane = w >> 4, i = w & 15;
    int kc = blk / 6, nt = blk % 6;
    int k = kc * 32 + ((lane < 16) ? 0 : 16) + i;
    int n = nt * 16 + (lane & 15);
    bool in = (k < 92) && (n < 92);
    we1b[f] = bfc(in ? We1[k * 92 + n] : 0.f);
    we2b[f] = bfc(in ? We2[k * 92 + n] : 0.f);
    float sv = 0.f;
    if (k < 92) sv = (n < NSC) ? Ws[k * NSC + n] : Wv[k * NVC + (n - NSC)];
    wsvb[f] = bfc(sv);
  }
  for (int f = tid; f < 2048; f += nth){          // Wr1: 1x4 blocks, K padded 8->32
    int blk = f >> 9, w = f & 511, lane = w >> 4, i = w & 15;
    int k = ((lane < 16) ? 0 : 16) + i;
    int n = blk * 16 + (lane & 15);
    wr1b[f] = bfc((k < 8) ? Wr1[k * 64 + n] : 0.f);
  }
  for (int f = tid; f < 6144; f += nth){          // Wr2: 2x6 blocks (64x96)
    int blk = f >> 9, w = f & 511, lane = w >> 4, i = w & 15;
    int kc = blk / 6, nt = blk % 6;
    int k = kc * 32 + ((lane < 16) ? 0 : 16) + i;
    int n = nt * 16 + (lane & 15);
    wr2b[f] = bfc(Wr2[k * 96 + n]);
  }
  for (int f = tid; f < 4096; f += nth){          // Wos: 2x4 blocks (64x64)
    int blk = f >> 9, w = f & 511, lane = w >> 4, i = w & 15;
    int kc = blk / 4, nt = blk % 4;
    int k = kc * 32 + ((lane < 16) ? 0 : 16) + i;
    int n = nt * 16 + (lane & 15);
    wosb[f] = bfc(Wos[k * 64 + n]);
  }
  for (int f = tid; f < 1024; f += nth){          // Wov: 1x2 blocks (32x32)
    int blk = f >> 9, w = f & 511, lane = w >> 4, i = w & 15;
    int k = ((lane < 16) ? 0 : 16) + i;
    int n = blk * 16 + (lane & 15);
    wovb[f] = bfc(Wov[k * 32 + n]);
  }
}

// ---------------- zero the atomic accumulators ----------------
__global__ void zero_f32(float* p, int n){
  int i = threadIdx.x + blockIdx.x * blockDim.x;
  const int s = blockDim.x * gridDim.x;
  for (; i < n; i += s) p[i] = 0.0f;
}

// ---------------- kernel 1: h = silu(x@We1)@We2 -> bf16 table ----------------
__global__ void __launch_bounds__(32)
node_mlp(const float* __restrict__ x, const __bf16* __restrict__ we1b,
         const __bf16* __restrict__ we2b, __bf16* __restrict__ hb){
  __shared__ __attribute__((aligned(16))) __bf16 xa[16 * 96];
  __shared__ __attribute__((aligned(16))) __bf16 t1[16 * 96];
  const int lane = threadIdx.x;
  const int n0 = blockIdx.x * 16;                 // 3125*16 == 50000 exactly
  for (int idx = lane; idx < 16 * 23; idx += 32){ // 92 floats = 23 x float4 per row
    int r = idx / 23, q = idx % 23;
    v4f v = *(const v4f*)(x + (size_t)(n0 + r) * 92 + q * 4);
    v4bf b; b[0] = bfc(v[0]); b[1] = bfc(v[1]); b[2] = bfc(v[2]); b[3] = bfc(v[3]);
    *(v4bf*)(xa + r * 96 + q * 4) = b;
  }
  if (lane < 16){                                 // zero pad cols 92..95
    v4bf zb = {(__bf16)0.f, (__bf16)0.f, (__bf16)0.f, (__bf16)0.f};
    *(v4bf*)(xa + lane * 96 + 92) = zb;
  }
  __asm__ volatile("" ::: "memory");              // per-wave LDS ops stay in order (DScnt)
  const int mb = (lane < 16) ? 0 : 8;
  const int cl = lane & 15;
  {                                               // GEMM1: A-frags hoisted (depend on kc only)
    v16bf a0 = load_afrag(xa, 96, 0);
    v16bf a1 = load_afrag(xa, 96, 32);
    v16bf a2 = load_afrag(xa, 96, 64);
#pragma unroll
    for (int nt = 0; nt < 6; ++nt){
      v8f acc = zero8();
      acc = wmma_bf16(a0, load_bfrag_sw(we1b + (0 * 6 + nt) * 512), acc);
      acc = wmma_bf16(a1, load_bfrag_sw(we1b + (1 * 6 + nt) * 512), acc);
      acc = wmma_bf16(a2, load_bfrag_sw(we1b + (2 * 6 + nt) * 512), acc);
#pragma unroll
      for (int j = 0; j < 8; ++j)
        t1[(mb + j) * 96 + nt * 16 + cl] = bfc(silu(acc[j]));
    }
  }
  __asm__ volatile("" ::: "memory");
  {                                               // GEMM2
    v16bf a0 = load_afrag(t1, 96, 0);
    v16bf a1 = load_afrag(t1, 96, 32);
    v16bf a2 = load_afrag(t1, 96, 64);
#pragma unroll
    for (int nt = 0; nt < 6; ++nt){
      v8f acc = zero8();
      acc = wmma_bf16(a0, load_bfrag_sw(we2b + (0 * 6 + nt) * 512), acc);
      acc = wmma_bf16(a1, load_bfrag_sw(we2b + (1 * 6 + nt) * 512), acc);
      acc = wmma_bf16(a2, load_bfrag_sw(we2b + (2 * 6 + nt) * 512), acc);
#pragma unroll
      for (int j = 0; j < 8; ++j)
        hb[(size_t)(n0 + mb + j) * 96 + nt * 16 + cl] = bfc(acc[j]);
    }
  }
}

// ---------------- kernel 2: fused radial MLP + edge GEMM + scatter ----------------
#define EK_WAVES 8
__global__ void __launch_bounds__(256)
edge_kernel(const __bf16* __restrict__ hb, const float* __restrict__ evec,
            const int* __restrict__ eidx,
            const __bf16* __restrict__ wedge,   // wr1|wr2|wsv swizzled, contiguous 17408
            float* __restrict__ agg_s, float* __restrict__ agg_v){
  __shared__ __attribute__((aligned(16))) __bf16 wl[17408];
  __shared__ __attribute__((aligned(16))) __bf16 hid[EK_WAVES][16 * 64];
  __shared__ __attribute__((aligned(16))) float  vh [EK_WAVES][16 * 3];
  __shared__ __attribute__((aligned(16))) int    dsti[EK_WAVES][16];
  const int tid = threadIdx.x;
  for (int i = tid; i < 17408 / 8; i += 256)      // vectorized weight stage (v8bf copies)
    ((v8bf*)wl)[i] = ((const v8bf*)wedge)[i];
  __syncthreads();
  const __bf16* Wr1 = wl;                         // 1x4 blocks
  const __bf16* Wr2 = wl + 2048;                  // 2x6 blocks
  const __bf16* Wsv = wl + 8192;                  // 3x6 blocks

  const int wid  = tid >> 5;
  const int lane = tid & 31;
  const int cl   = lane & 15;
  const int mb   = (lane < 16) ? 0 : 8;
  const int ntiles = N_EDGES / 16;                // 62500 exactly

  for (int t = blockIdx.x * EK_WAVES + wid; t < ntiles; t += gridDim.x * EK_WAVES){
    const int e0 = t * 16;
    // ---- radial basis: A-frag of rb needs no staging (lanes<16 hold K=0..7 of own row) ----
    v16bf arb;
#pragma unroll
    for (int i = 0; i < 16; ++i) arb[i] = (__bf16)0.0f;
    if (lane < 16){
      const int e = e0 + lane;
      float vx = evec[e * 3 + 0], vy = evec[e * 3 + 1], vz = evec[e * 3 + 2];
      float r = sqrtf(vx * vx + vy * vy + vz * vz + EPSR);
      float ri = 1.0f / r;
      vh[wid][lane * 3 + 0] = vx * ri;
      vh[wid][lane * 3 + 1] = vy * ri;
      vh[wid][lane * 3 + 2] = vz * ri;
      dsti[wid][lane] = eidx[e];                  // dst = edge_index[0]
      float u = r * 0.2f;                         // r / RMAX
      float u2 = u * u, u3 = u2 * u, u6 = u3 * u3, u7 = u6 * u, u8 = u7 * u;
      float fc = (u < 1.0f) ? (1.0f - 28.0f * u6 + 48.0f * u7 - 21.0f * u8) : 0.0f;
      float pref = 0.6324555320336759f * ri * fc; // sqrt(2/RMAX)
      float th = r * 0.6283185307179586f;         // pi/RMAX
#pragma unroll
      for (int k = 0; k < 8; ++k)
        arb[k] = bfc(pref * __sinf((float)(k + 1) * th));
    }
    __asm__ volatile("" ::: "memory");
    // ---- hidden = silu(rb @ Wr1) : rows 8..31 of Wr1 are zero-padded ----
#pragma unroll
    for (int nt = 0; nt < 4; ++nt){
      v8f acc = zero8();
      acc = wmma_bf16(arb, load_bfrag_sw(Wr1 + nt * 512), acc);
#pragma unroll
      for (int j = 0; j < 8; ++j)
        hid[wid][(mb + j) * 64 + nt * 16 + cl] = bfc(silu(acc[j]));
    }
    __asm__ volatile("" ::: "memory");
    // ---- A-frags: hidden (LDS b128 loads) and gathered h[src] (global b128, L2-resident) ----
    v16bf ah0 = load_afrag(hid[wid], 64, 0);
    v16bf ah1 = load_afrag(hid[wid], 64, 32);
    const int s = eidx[N_EDGES + e0 + cl];        // src = edge_index[1]
    const __bf16* hp = hb + (size_t)s * 96 + ((lane < 16) ? 0 : 16);
    v16bf ax[3];
#pragma unroll
    for (int kc = 0; kc < 3; ++kc)
      ax[kc] = concat8(*(const v8bf*)(hp + kc * 32), *(const v8bf*)(hp + kc * 32 + 8));
    // ---- wgt (16x96) and x@[Ws|Wv] (16x96) in identical D layouts -> register multiply ----
    v8f res[6];
#pragma unroll
    for (int nt = 0; nt < 6; ++nt){
      v8f wacc = zero8();
      wacc = wmma_bf16(ah0, load_bfrag_sw(Wr2 + (0 * 6 + nt) * 512), wacc);
      wacc = wmma_bf16(ah1, load_bfrag_sw(Wr2 + (1 * 6 + nt) * 512), wacc);
      v8f xacc = zero8();
      xacc = wmma_bf16(ax[0], load_bfrag_sw(Wsv + (0 * 6 + nt) * 512), xacc);
      xacc = wmma_bf16(ax[1], load_bfrag_sw(Wsv + (1 * 6 + nt) * 512), xacc);
      xacc = wmma_bf16(ax[2], load_bfrag_sw(Wsv + (2 * 6 + nt) * 512), xacc);
#pragma unroll
      for (int j = 0; j < 8; ++j) res[nt][j] = wacc[j] * xacc[j] * INV_DEG;
    }
    // ---- scatter-add (L2-resident accumulators); vectorized LDS reads of dst/vhat ----
    v4i dlo = *(const v4i*)(&dsti[wid][mb]);
    v4i dhi = *(const v4i*)(&dsti[wid][mb + 4]);
    int dn[8] = { dlo[0], dlo[1], dlo[2], dlo[3], dhi[0], dhi[1], dhi[2], dhi[3] };
    v4f vb0 = *(const v4f*)(&vh[wid][mb * 3 + 0]);   // rows mb..mb+7: 24 contiguous floats
    v4f vb1 = *(const v4f*)(&vh[wid][mb * 3 + 4]);
    v4f vb2 = *(const v4f*)(&vh[wid][mb * 3 + 8]);
    v4f vb3 = *(const v4f*)(&vh[wid][mb * 3 + 12]);
    v4f vb4 = *(const v4f*)(&vh[wid][mb * 3 + 16]);
    v4f vb5 = *(const v4f*)(&vh[wid][mb * 3 + 20]);
    float vhl[24] = { vb0[0],vb0[1],vb0[2],vb0[3], vb1[0],vb1[1],vb1[2],vb1[3],
                      vb2[0],vb2[1],vb2[2],vb2[3], vb3[0],vb3[1],vb3[2],vb3[3],
                      vb4[0],vb4[1],vb4[2],vb4[3], vb5[0],vb5[1],vb5[2],vb5[3] };
#pragma unroll
    for (int nt = 0; nt < 4; ++nt){               // scalar messages: cols 0..63
      const int col = nt * 16 + cl;
#pragma unroll
      for (int j = 0; j < 8; ++j)
        atomicAdd(&agg_s[(size_t)dn[j] * NSC + col], res[nt][j]);
    }
#pragma unroll
    for (int nt = 4; nt < 6; ++nt){               // vector messages: channel c in 0..31
      const int c = (nt - 4) * 16 + cl;
#pragma unroll
      for (int j = 0; j < 8; ++j){
        const float v = res[nt][j];
        float* b = &agg_v[(size_t)dn[j] * (NVC * 3) + c * 3];
        atomicAdd(b + 0, v * vhl[j * 3 + 0]);
        atomicAdd(b + 1, v * vhl[j * 3 + 1]);
        atomicAdd(b + 2, v * vhl[j * 3 + 2]);
      }
    }
  }
}

// ---------------- kernel 3: output GEMMs + silu ----------------
__global__ void __launch_bounds__(32)
out_kernel(const float* __restrict__ agg_s, const float* __restrict__ agg_v,
           const __bf16* __restrict__ wosb, const __bf16* __restrict__ wovb,
           float* __restrict__ out){
  __shared__ __attribute__((aligned(16))) __bf16 as[16 * 64];
  __shared__ __attribute__((aligned(16))) __bf16 av[3][16 * 32];
  const int lane = threadIdx.x;
  const int n0 = blockIdx.x * 16;
  for (int idx = lane; idx < 16 * 16; idx += 32){ // agg_s tile: float4 in, v4bf out
    int r = idx >> 4, q = idx & 15;
    v4f v = *(const v4f*)(agg_s + (size_t)(n0 + r) * NSC + q * 4);
    v4bf b; b[0] = bfc(v[0]); b[1] = bfc(v[1]); b[2] = bfc(v[2]); b[3] = bfc(v[3]);
    *(v4bf*)(as + r * 64 + q * 4) = b;
  }
  for (int idx = lane; idx < 16 * 24; idx += 32){ // agg_v row = 24 x float4, de-interleave (c,d)
    int r = idx / 24, q = idx % 24;
    v4f v = *(const v4f*)(agg_v + (size_t)(n0 + r) * (NVC * 3) + q * 4);
#pragma unroll
    for (int tt = 0; tt < 4; ++tt){
      int flat = q * 4 + tt, d = flat % 3, c = flat / 3;
      av[d][r * 32 + c] = bfc(v[tt]);
    }
  }
  __asm__ volatile("" ::: "memory");
  const int mb = (lane < 16) ? 0 : 8;
  const int cl = lane & 15;
  {                                               // out_s = agg_s @ W_os (A-frags hoisted)
    v16bf a0 = load_afrag(as, 64, 0);
    v16bf a1 = load_afrag(as, 64, 32);
#pragma unroll
    for (int nt = 0; nt < 4; ++nt){
      v8f acc = zero8();
      acc = wmma_bf16(a0, load_bfrag_sw(wosb + (0 * 4 + nt) * 512), acc);
      acc = wmma_bf16(a1, load_bfrag_sw(wosb + (1 * 4 + nt) * 512), acc);
#pragma unroll
      for (int j = 0; j < 8; ++j)
        out[(size_t)(n0 + mb + j) * 160 + nt * 16 + cl] = silu(acc[j]);
    }
  }
#pragma unroll
  for (int d = 0; d < 3; ++d){                    // out_v[n,k,d] = sum_c agg_v[n,c,d] W_ov[c,k]
    v16bf a = load_afrag(av[d], 32, 0);
#pragma unroll
    for (int nt = 0; nt < 2; ++nt){
      v8f acc = zero8();
      acc = wmma_bf16(a, load_bfrag_sw(wovb + nt * 512), acc);
#pragma unroll
      for (int j = 0; j < 8; ++j)
        out[(size_t)(n0 + mb + j) * 160 + 64 + (nt * 16 + cl) * 3 + d] = silu(acc[j]);
    }
  }
}

extern "C" void kernel_launch(void* const* d_in, const int* in_sizes, int n_in,
                              void* d_out, int out_size, void* d_ws, size_t ws_size,
                              hipStream_t stream){
  (void)in_sizes; (void)n_in; (void)out_size; (void)ws_size;
  const float* x    = (const float*)d_in[0];
  const float* evec = (const float*)d_in[1];
  const int*   eidx = (const int*)d_in[2];
  const float* We1  = (const float*)d_in[3];
  const float* We2  = (const float*)d_in[4];
  const float* Wr1  = (const float*)d_in[5];
  const float* Wr2  = (const float*)d_in[6];
  const float* Ws   = (const float*)d_in[7];
  const float* Wv   = (const float*)d_in[8];
  const float* Wos  = (const float*)d_in[9];
  const float* Wov  = (const float*)d_in[10];

  char* ws = (char*)d_ws;
  __bf16* hb   = (__bf16*)(ws);                   //  9,600,000 B : bf16 h table (L2-resident)
  float*  aggs = (float*)(ws + 9600000);          // 12,800,000 B : agg_s
  float*  aggv = (float*)(ws + 22400000);         // 19,200,000 B : agg_v (contiguous w/ agg_s)
  __bf16* we1b = (__bf16*)(ws + 41600000);        // swizzled weights:
  __bf16* we2b = we1b + 9216;
  __bf16* wosb = we2b + 9216;
  __bf16* wovb = wosb + 4096;
  __bf16* wr1b = wovb + 1024;                     // wr1|wr2|wsv contiguous (17408) for the
  __bf16* wr2b = wr1b + 2048;                     // edge kernel's single LDS stage
  __bf16* wsvb = wr2b + 6144;

  prep_weights<<<32, 256, 0, stream>>>(We1, We2, Wr1, Wr2, Ws, Wv, Wos, Wov,
                                       we1b, we2b, wosb, wovb, wr1b, wr2b, wsvb);
  zero_f32<<<2048, 256, 0, stream>>>(aggs, N_NODES * (NSC + NVC * 3));
  node_mlp<<<N_NODES / 16, 32, 0, stream>>>(x, we1b, we2b, hb);
  edge_kernel<<<1024, 256, 0, stream>>>(hb, evec, eidx, wr1b, aggs, aggv);
  out_kernel<<<N_NODES / 16, 32, 0, stream>>>(aggs, aggv, wosb, wovb, (float*)d_out);
}
